// MultiHeadRelevantTokenAttention_20864951124445
// MI455X (gfx1250) — matrix operand order repaired and checked by reading, not compiled, not run on hardware
//
#include <hip/hip_runtime.h>

// ---------------------------------------------------------------------------
// MultiHeadRelevantTokenAttention for MI455X (gfx1250, wave32).
//
// Algebraic restructuring (see analysis): K/V projections are folded away;
// the only surviving GEMM is scores = (q~ * g / 8) @ x^T per batch, done with
// v_wmma_f32_16x16x32_f16 (8 heads padded to 16 rows). LayerNorm is folded
// into the dot products via per-token mean/rstd. Total HBM traffic ~2 reads
// of x (512 MB) -> ~22us roofline at 23.3 TB/s.
// ---------------------------------------------------------------------------

#define BB   64
#define NN   1024
#define DD   1024
#define QKV_ 512
#define HH   8
#define HDD  64
#define TK   32
#define LNEPS 1e-5f

typedef __attribute__((ext_vector_type(16))) _Float16 v16h;
typedef __attribute__((ext_vector_type(8)))  _Float16 h8;
typedef __attribute__((ext_vector_type(8)))  float    v8f;

// --- K1: per-token mean / rstd / scorer logit (one wave per token) ---------
__global__ void k_token_stats(const float* __restrict__ x,
                              const float* __restrict__ sw,
                              const float* __restrict__ sb,
                              float* __restrict__ logits,
                              float* __restrict__ mean,
                              float* __restrict__ rstd) {
  int wid = threadIdx.x >> 5, lane = threadIdx.x & 31;
  int tok = blockIdx.x * 8 + wid;            // grid = B*N/8
  const float* row = x + (size_t)tok * DD;
  float s = 0.f, ss = 0.f, dt = 0.f;
  for (int i = 0; i < DD / 32; i++) {
    float v = row[lane + i * 32];
    s += v; ss += v * v; dt += v * sw[lane + i * 32];
  }
  for (int o = 16; o; o >>= 1) {
    s  += __shfl_xor(s, o);
    ss += __shfl_xor(ss, o);
    dt += __shfl_xor(dt, o);
  }
  if (lane == 0) {
    float m = s * (1.f / DD);
    float var = ss * (1.f / DD) - m * m;
    mean[tok] = m;
    rstd[tok] = rsqrtf(var + LNEPS);
    logits[tok] = dt + sb[0];
  }
}

// --- K2: argmax over N logits per batch (ties -> lowest index) -------------
__global__ void k_argmax(const float* __restrict__ logits, int* __restrict__ sel) {
  __shared__ float sv[256];
  __shared__ int   si[256];
  int b = blockIdx.x, t = threadIdx.x;
  float bv = -INFINITY; int bi = NN;
  for (int n = t; n < NN; n += 256) {
    float v = logits[b * NN + n];
    if (v > bv || (v == bv && n < bi)) { bv = v; bi = n; }
  }
  sv[t] = bv; si[t] = bi; __syncthreads();
  for (int o = 128; o; o >>= 1) {
    if (t < o) {
      float v = sv[t + o]; int i = si[t + o];
      if (v > sv[t] || (v == sv[t] && i < si[t])) { sv[t] = v; si[t] = i; }
    }
    __syncthreads();
  }
  if (t == 0) sel[b] = si[0];
}

// --- K3: selected token: s_n = LN(sel_tok); q = s_n@qk_w^T+qk_b;
//         skip = sel_tok@skip_w^T+skip_b; bias gather -----------------------
__global__ void k_selected(const float* __restrict__ x,
                           const float* __restrict__ mean,
                           const float* __restrict__ rstd,
                           const int*   __restrict__ sel,
                           const float* __restrict__ g,
                           const float* __restrict__ lb,
                           const float* __restrict__ qkw,
                           const float* __restrict__ qkb,
                           const float* __restrict__ skw,
                           const float* __restrict__ skb,
                           const float* __restrict__ ubias,
                           const int*   __restrict__ relidx,
                           float* __restrict__ q,
                           float* __restrict__ skip,
                           float* __restrict__ biassel) {
  __shared__ float sn[DD];
  __shared__ float raw[DD];
  int b = blockIdx.x, t = threadIdx.x;
  int s = sel[b];
  float m = mean[b * NN + s], r = rstd[b * NN + s];
  const float* row = x + ((size_t)b * NN + s) * DD;
  for (int i = t; i < DD; i += 256) {
    float v = row[i];
    raw[i] = v;
    sn[i] = (v - m) * r * g[i] + lb[i];
  }
  for (int n = t; n < NN; n += 256)
    biassel[b * NN + n] = ubias[relidx[s * NN + n]];
  __syncthreads();
  for (int j = t; j < QKV_; j += 256) {
    const float* wr  = qkw + (size_t)j * DD;
    const float* wr2 = skw + (size_t)j * DD;
    float a = 0.f, c = 0.f;
    for (int i = 0; i < DD; i++) { a += sn[i] * wr[i]; c += raw[i] * wr2[i]; }
    q[b * QKV_ + j]    = a + qkb[j];
    skip[b * QKV_ + j] = c + skb[j];
  }
}

// --- K4a: zero padding rows 8..15 of the f16 A matrix ----------------------
__global__ void k_zero_pad(_Float16* __restrict__ Amat) {
  int b = blockIdx.x, t = threadIdx.x;
  _Float16* base = Amat + ((size_t)b * 16 + 8) * DD;
  for (int i = t; i < 8 * DD; i += 256) base[i] = (_Float16)0.f;
}

// --- K4: q~[b,h,t] = sum_d q[b,h,d]*qk_w[h*64+d,t]; store A = q~*g/8 (f16);
//         S1 = sum(q~*g)/8; base = (sum(q~*lnb) + q.qk_b)/8 -----------------
__global__ void k_qtilde(const float* __restrict__ q,
                         const float* __restrict__ qkw,
                         const float* __restrict__ qkb,
                         const float* __restrict__ g,
                         const float* __restrict__ lb,
                         _Float16* __restrict__ Amat,
                         float* __restrict__ S1,
                         float* __restrict__ base) {
  __shared__ float qh[HDD];
  __shared__ float red[256];
  __shared__ float red2[256];
  int bh = blockIdx.x, b = bh / HH, h = bh % HH, t = threadIdx.x;
  if (t < HDD) qh[t] = q[b * QKV_ + h * HDD + t];
  __syncthreads();
  float s1 = 0.f, s2 = 0.f;
  for (int i = 0; i < DD / 256; i++) {
    int tt = t + i * 256;
    float qt = 0.f;
    for (int d = 0; d < HDD; d++) qt += qh[d] * qkw[(size_t)(h * HDD + d) * DD + tt];
    float a = qt * g[tt] * 0.125f;
    Amat[((size_t)b * 16 + h) * DD + tt] = (_Float16)a;
    s1 += a;
    s2 += qt * lb[tt] * 0.125f;
  }
  red[t] = s1; red2[t] = s2; __syncthreads();
  for (int o = 128; o; o >>= 1) {
    if (t < o) { red[t] += red[t + o]; red2[t] += red2[t + o]; }
    __syncthreads();
  }
  if (t == 0) {
    float c = 0.f;
    for (int d = 0; d < HDD; d++) c += qh[d] * qkb[h * HDD + d];
    S1[bh]   = red[0];
    base[bh] = red2[0] + c * 0.125f;
  }
}

// --- K5: WMMA score GEMM: D16x16 = A(16 heads x K) * X^T(K x 16 tokens) ----
// One 16-token tile per wave; 8 waves/block; k-loop of 32 with f16 WMMA.
__global__ void k_attn_wmma(const float* __restrict__ x,
                            const _Float16* __restrict__ Amat,
                            const float* __restrict__ mean,
                            const float* __restrict__ rstd,
                            const float* __restrict__ S1,
                            const float* __restrict__ base,
                            const float* __restrict__ biassel,
                            float* __restrict__ attn) {
  int wid = threadIdx.x >> 5, lane = threadIdx.x & 31;
  int b = blockIdx.x >> 3;
  int tile = (blockIdx.x & 7) * 8 + wid;
  int n0 = tile * 16;
  int row16 = lane & 15, hi = lane >> 4;
  // A fragment source: row m = row16 of this batch's 16x1024 f16 matrix.
  const _Float16* Ab = Amat + ((size_t)b * 16 + row16) * DD;
  // B fragment source: lane's token column (same token for hi=0/1 halves).
  int tokn = n0 + row16;
  const float* xrow = x + ((size_t)b * NN + tokn) * DD;
  v8f c = {};
  for (int k = 0; k < DD; k += 32) {
    // A 16x32 f16 layout: lanes<16 hold K {k..k+7, k+16..k+23},
    // lanes>=16 hold K {k+8..k+15, k+24..k+31} (ISA 7.12.2).
    h8 a0 = *(const h8*)(Ab + k + hi * 8);
    h8 a1 = *(const h8*)(Ab + k + 16 + hi * 8);
    v16h a;
#pragma unroll
    for (int j = 0; j < 8; j++) { a[j] = a0[j]; a[8 + j] = a1[j]; }
    // B 32x16 f16 layout: lane = column N, K same as A: {k+8hi..+7, k+16+8hi..+7}.
    v16h bb;
#pragma unroll
    for (int j = 0; j < 4; j++) {
      float4 f = *(const float4*)(xrow + k + hi * 8 + (j >> 1) * 16 + (j & 1) * 4);
      bb[4 * j + 0] = (_Float16)f.x;
      bb[4 * j + 1] = (_Float16)f.y;
      bb[4 * j + 2] = (_Float16)f.z;
      bb[4 * j + 3] = (_Float16)f.w;
    }
    if (k + 32 < DD) __builtin_prefetch(xrow + k + 32, 0, 1);
    c = __builtin_amdgcn_wmma_f32_16x16x32_f16(false, a, false, bb,
                                               (short)0, c, false, false);
  }
  // Lanes 0..15 hold heads 0..7 in c[0..7] for token n0+lane (M=j, N=lane).
  if (hi == 0) {
    float r = rstd[b * NN + tokn];
    float m = mean[b * NN + tokn];
    float bs = biassel[b * NN + tokn];
#pragma unroll
    for (int j = 0; j < HH; j++) {
      float sc = r * c[j] - r * m * S1[b * HH + j] + base[b * HH + j] + bs;
      attn[((size_t)(b * HH + j)) * NN + tokn] = sc;
    }
  }
}

// --- K6: softmax over N, iterative top-32, renormalize ---------------------
__global__ void k_softmax_topk(const float* __restrict__ attn,
                               int* __restrict__ tidx,
                               float* __restrict__ tw) {
  __shared__ float p[NN];
  __shared__ float red[256];
  __shared__ int   ired[256];
  __shared__ float tv[TK];
  __shared__ int   ti[TK];
  __shared__ float wsum_s;
  int bh = blockIdx.x, t = threadIdx.x;
  const float* s = attn + (size_t)bh * NN;
  float mx = -INFINITY;
  for (int n = t; n < NN; n += 256) { float v = s[n]; p[n] = v; mx = fmaxf(mx, v); }
  red[t] = mx; __syncthreads();
  for (int o = 128; o; o >>= 1) { if (t < o) red[t] = fmaxf(red[t], red[t + o]); __syncthreads(); }
  mx = red[0]; __syncthreads();
  float sum = 0.f;
  for (int n = t; n < NN; n += 256) { float e = __expf(p[n] - mx); p[n] = e; sum += e; }
  red[t] = sum; __syncthreads();
  for (int o = 128; o; o >>= 1) { if (t < o) red[t] += red[t + o]; __syncthreads(); }
  float inv = 1.f / red[0];
  __syncthreads();
  for (int n = t; n < NN; n += 256) p[n] *= inv;
  if (t == 0) wsum_s = 0.f;
  __syncthreads();
  for (int k = 0; k < TK; k++) {
    float bv = -1.f; int bi = NN;
    for (int n = t; n < NN; n += 256) {
      float v = p[n];
      if (v > bv || (v == bv && n < bi)) { bv = v; bi = n; }
    }
    red[t] = bv; ired[t] = bi; __syncthreads();
    for (int o = 128; o; o >>= 1) {
      if (t < o) {
        float v = red[t + o]; int i = ired[t + o];
        if (v > red[t] || (v == red[t] && i < ired[t])) { red[t] = v; ired[t] = i; }
      }
      __syncthreads();
    }
    if (t == 0) { tv[k] = red[0]; ti[k] = ired[0]; p[ired[0]] = -1.f; wsum_s += red[0]; }
    __syncthreads();
  }
  if (t < TK) {
    tw[bh * TK + t]   = tv[t] / (wsum_s + 1e-9f);
    tidx[bh * TK + t] = ti[t];
  }
}

// --- K7: z = g*(sum_k w_k r_k x_nk - c0) + lnb ; y = v_w_h @ z + v_b -------
__global__ void k_gather_value(const float* __restrict__ x,
                               const float* __restrict__ mean,
                               const float* __restrict__ rstd,
                               const int*   __restrict__ tidx,
                               const float* __restrict__ tw,
                               const float* __restrict__ g,
                               const float* __restrict__ lb,
                               const float* __restrict__ vw,
                               const float* __restrict__ vb,
                               float* __restrict__ ybuf) {
  __shared__ float wk[TK];
  __shared__ int   ti[TK];
  __shared__ float c0s;
  __shared__ float z[DD];
  __shared__ float red[256];
  int bh = blockIdx.x, b = bh / HH, h = bh % HH, t = threadIdx.x;
  if (t < TK) {
    int idx = tidx[bh * TK + t];
    float w = tw[bh * TK + t];
    float r = rstd[b * NN + idx];
    wk[t] = w * r;
    ti[t] = idx;
    red[t] = w * r * mean[b * NN + idx];
  }
  __syncthreads();
  if (t == 0) { float c = 0.f; for (int i = 0; i < TK; i++) c += red[i]; c0s = c; }
  __syncthreads();
  float c0 = c0s;
  for (int i = 0; i < DD / 256; i++) {
    int tt = t + i * 256;
    float u = 0.f;
    for (int k = 0; k < TK; k++) u += wk[k] * x[((size_t)b * NN + ti[k]) * DD + tt];
    z[tt] = g[tt] * (u - c0) + lb[tt];
  }
  __syncthreads();
  int d = t & 63, qtr = t >> 6;
  const float* wr = vw + (size_t)(h * HDD + d) * DD + qtr * 256;
  const float* zz = z + qtr * 256;
  float pp = 0.f;
  for (int i = 0; i < 256; i++) pp += wr[i] * zz[i];
  red[t] = pp; __syncthreads();
  if (t < HDD)
    ybuf[(size_t)b * QKV_ + h * HDD + t] =
        red[t] + red[t + 64] + red[t + 128] + red[t + 192] + vb[h * HDD + t];
}

// --- K8: relu(y+skip) -> out_w proj -> fc --------------------------------
__global__ void k_head(const float* __restrict__ ybuf,
                       const float* __restrict__ skip,
                       const float* __restrict__ ow,
                       const float* __restrict__ ob,
                       const float* __restrict__ fw,
                       const float* __restrict__ fb,
                       float* __restrict__ out) {
  __shared__ float yv[QKV_];
  __shared__ float y2[QKV_];
  __shared__ float r0[256];
  __shared__ float r1[256];
  int b = blockIdx.x, t = threadIdx.x;
  for (int j = t; j < QKV_; j += 256)
    yv[j] = fmaxf(ybuf[b * QKV_ + j] + skip[b * QKV_ + j], 0.f);
  __syncthreads();
  for (int j = t; j < QKV_; j += 256) {
    const float* wr = ow + (size_t)j * QKV_;
    float a = 0.f;
    for (int i = 0; i < QKV_; i++) a += yv[i] * wr[i];
    y2[j] = a + ob[j];
  }
  __syncthreads();
  r0[t] = y2[t] * fw[t]         + y2[t + 256] * fw[t + 256];
  r1[t] = y2[t] * fw[QKV_ + t]  + y2[t + 256] * fw[QKV_ + t + 256];
  __syncthreads();
  for (int o = 128; o; o >>= 1) {
    if (t < o) { r0[t] += r0[t + o]; r1[t] += r1[t + o]; }
    __syncthreads();
  }
  if (t == 0) {
    out[b * 2 + 0] = r0[0] + fb[0];
    out[b * 2 + 1] = r1[0] + fb[1];
  }
}

extern "C" void kernel_launch(void* const* d_in, const int* in_sizes, int n_in,
                              void* d_out, int out_size, void* d_ws, size_t ws_size,
                              hipStream_t stream) {
  const float* x        = (const float*)d_in[0];
  const float* scorer_w = (const float*)d_in[1];
  const float* scorer_b = (const float*)d_in[2];
  const float* ln_g     = (const float*)d_in[3];
  const float* ln_b     = (const float*)d_in[4];
  const float* qk_w     = (const float*)d_in[5];
  const float* qk_b     = (const float*)d_in[6];
  const float* v_w      = (const float*)d_in[7];
  const float* v_b      = (const float*)d_in[8];
  const float* skip_w   = (const float*)d_in[9];
  const float* skip_b   = (const float*)d_in[10];
  const float* out_w    = (const float*)d_in[11];
  const float* out_b    = (const float*)d_in[12];
  const float* fc_w     = (const float*)d_in[13];
  const float* fc_b     = (const float*)d_in[14];
  const float* ubias    = (const float*)d_in[15];
  const int*   relidx   = (const int*)d_in[16];
  float* out = (float*)d_out;

  char* ws = (char*)d_ws;
  size_t off = 0;
  auto alloc = [&](size_t bytes) -> void* {
    void* p = ws + off;
    off = (off + bytes + 255) & ~(size_t)255;
    return p;
  };
  float*    logits  = (float*)alloc((size_t)BB * NN * 4);
  float*    mean    = (float*)alloc((size_t)BB * NN * 4);
  float*    rstd    = (float*)alloc((size_t)BB * NN * 4);
  int*      sel     = (int*)  alloc((size_t)BB * 4);
  float*    q       = (float*)alloc((size_t)BB * QKV_ * 4);
  float*    skip    = (float*)alloc((size_t)BB * QKV_ * 4);
  float*    biassel = (float*)alloc((size_t)BB * NN * 4);
  float*    S1      = (float*)alloc((size_t)BB * HH * 4);
  float*    base    = (float*)alloc((size_t)BB * HH * 4);
  _Float16* Amat    = (_Float16*)alloc((size_t)BB * 16 * DD * 2);
  float*    attn    = (float*)alloc((size_t)BB * HH * NN * 4);
  int*      tidx    = (int*)  alloc((size_t)BB * HH * TK * 4);
  float*    tw      = (float*)alloc((size_t)BB * HH * TK * 4);
  float*    ybuf    = (float*)alloc((size_t)BB * QKV_ * 4);

  k_token_stats<<<BB * NN / 8, 256, 0, stream>>>(x, scorer_w, scorer_b,
                                                 logits, mean, rstd);
  k_argmax<<<BB, 256, 0, stream>>>(logits, sel);
  k_selected<<<BB, 256, 0, stream>>>(x, mean, rstd, sel, ln_g, ln_b,
                                     qk_w, qk_b, skip_w, skip_b,
                                     ubias, relidx, q, skip, biassel);
  k_zero_pad<<<BB, 256, 0, stream>>>(Amat);
  k_qtilde<<<BB * HH, 256, 0, stream>>>(q, qk_w, qk_b, ln_g, ln_b,
                                        Amat, S1, base);
  k_attn_wmma<<<BB * 8, 256, 0, stream>>>(x, Amat, mean, rstd, S1, base,
                                          biassel, attn);
  k_softmax_topk<<<BB * HH, 256, 0, stream>>>(attn, tidx, tw);
  k_gather_value<<<BB * HH, 256, 0, stream>>>(x, mean, rstd, tidx, tw,
                                              ln_g, ln_b, v_w, v_b, ybuf);
  k_head<<<BB, 256, 0, stream>>>(ybuf, skip, out_w, out_b, fc_w, fc_b, out);
}
